// StyleGAN2GeneratorClean_74637941670071
// MI455X (gfx1250) — matrix operand order repaired and checked by reading, not compiled, not run on hardware
//
#include <hip/hip_runtime.h>
#include <hip/hip_bf16.h>
#include <math.h>

typedef __attribute__((ext_vector_type(16))) _Float16 v16h;
typedef __attribute__((ext_vector_type(8)))  float    v8f;

#define NSF   512
#define BATCH 8
#define CONV_WAVES 8   // waves per modconv block -> 128 output channels per block

// gfx1250 async global->LDS copy, guarded so the build never breaks
#if defined(__has_builtin)
#  if __has_builtin(__builtin_amdgcn_global_load_async_to_lds_b32)
#    define SG2_ASYNC_LDS 1
#  endif
#endif
#ifndef SG2_ASYNC_LDS
#  define SG2_ASYNC_LDS 0
#endif

typedef __attribute__((address_space(1))) int as1_int;   // global (AS1) word ptr
typedef __attribute__((address_space(3))) int as3_int;   // LDS (AS3) word ptr

// ---- WMMA fragment K-index helper (CDNA5 ISA §7.12.2, wave32) ----
// A matrix 16x32 f16, lane row m = lane&15, half hi = lane>>4:
// element e holds local K = (e&7) + 8*hi + 16*(e>>3)   (two contiguous runs of 8)
__device__ __forceinline__ int a_kidx(int e, int hi) {
    return (e & 7) + (hi << 3) + ((e >> 3) << 4);
}
// B matrix 32x16 f16: element e uses A's K order (MLP: a_kidx; conv/toRGB: e + 16*hi), column n = lane&15.

// ---- bilinear 2x upsample tap (half-pixel, edge clamp; matches jax.image.resize) ----
__device__ __forceinline__ float bilin2x(const float* __restrict__ plane,
                                         int inH, int inW, int oy, int ox) {
    int r0, r1; float wy0, wy1;
    if (oy & 1) { r0 = oy >> 1; r1 = r0 + 1; wy0 = 0.75f; wy1 = 0.25f; }
    else        { r1 = oy >> 1; r0 = r1 - 1; wy0 = 0.25f; wy1 = 0.75f; }
    if (r0 < 0) r0 = 0;
    if (r1 > inH - 1) r1 = inH - 1;
    int c0, c1; float wx0, wx1;
    if (ox & 1) { c0 = ox >> 1; c1 = c0 + 1; wx0 = 0.75f; wx1 = 0.25f; }
    else        { c1 = ox >> 1; c0 = c1 - 1; wx0 = 0.25f; wx1 = 0.75f; }
    if (c0 < 0) c0 = 0;
    if (c1 > inW - 1) c1 = inW - 1;
    const float* p0 = plane + (size_t)r0 * inW;
    const float* p1 = plane + (size_t)r1 * inW;
    return wy0 * (wx0 * p0[c0] + wx1 * p0[c1]) +
           wy1 * (wx0 * p1[c0] + wx1 * p1[c1]);
}

// ---- RMS-norm + 8-layer style MLP, resident in LDS, WMMA per 16-col tile ----
__global__ __launch_bounds__(1024)
void sg2_style_mlp_kernel(const float* __restrict__ styles,
                          const float* __restrict__ mlp_w,
                          const float* __restrict__ mlp_b,
                          float* __restrict__ latent) {
    __shared__ float buf0[BATCH * NSF];
    __shared__ float buf1[BATCH * NSF];
    __shared__ float ssum[BATCH];
    const int tid  = threadIdx.x;
    const int lane = tid & 31;
    const int wave = tid >> 5;          // 32 waves -> 32 column tiles of 16 = 512
    const int m    = lane & 15;
    const int hi   = lane >> 4;
    const float rowscale = (m < BATCH) ? 1.f : 0.f;  // rows 8..15 of A are zero

    if (tid < BATCH) ssum[tid] = 0.f;
    __syncthreads();
    for (int i = tid; i < BATCH * NSF; i += 1024) {
        float v = styles[i];
        atomicAdd(&ssum[i >> 9], v * v);     // ds_add_f32
    }
    __syncthreads();
    for (int i = tid; i < BATCH * NSF; i += 1024)
        buf0[i] = styles[i] * rsqrtf(ssum[i >> 9] * (1.f / NSF) + 1e-8f);
    __syncthreads();

    float* cur = buf0;
    float* nxt = buf1;
    const int nbase = wave << 4;
    const int mrow  = (m & 7) * NSF;     // clamped row: unconditional LDS load
    for (int l = 0; l < 8; ++l) {
        const float* W  = mlp_w + (size_t)l * NSF * NSF;   // [out=512][in=512]
        const float* bi = mlp_b + l * NSF;
        v8f c = {};
        for (int kb = 0; kb < NSF; kb += 32) {
            v16h a, bf;
#pragma unroll
            for (int e = 0; e < 16; ++e) {
                int ka = kb + a_kidx(e, hi);
                a[e] = (_Float16)(cur[mrow + ka] * rowscale);
                int kx = kb + a_kidx(e, hi);
                bf[e] = (_Float16)W[(size_t)(nbase + m) * NSF + kx];
            }
            c = __builtin_amdgcn_wmma_f32_16x16x32_f16(false, a, false, bf,
                                                       (short)0, c, false, false);
        }
        if (hi == 0) {   // rows 0..7 = batch
#pragma unroll
            for (int r = 0; r < 8; ++r) {
                float v = c[r] + bi[nbase + m];
                v = (v > 0.f) ? v : 0.2f * v;
                nxt[r * NSF + nbase + m] = v;
            }
        }
        __syncthreads();
        float* t = cur; cur = nxt; nxt = t;
    }
    for (int i = tid; i < BATCH * NSF; i += 1024) latent[i] = cur[i];
}

// ---- per-sample modulation vector: s_mod[b,ci] = latent[b]·mod_w[ci] + mod_b[ci] ----
__global__ void sg2_smod_kernel(const float* __restrict__ latent,
                                const float* __restrict__ mod_w,
                                const float* __restrict__ mod_b,
                                float* __restrict__ s_mod, int cin) {
    int i = blockIdx.x * blockDim.x + threadIdx.x;
    if (i >= BATCH * cin) return;
    int b = i / cin, ci = i - b * cin;
    const float* lv = latent + b * NSF;
    const float* mw = mod_w + (size_t)ci * NSF;
    float acc = 0.f;
#pragma unroll 4
    for (int k = 0; k < NSF; ++k) acc += lv[k] * mw[k];
    s_mod[i] = acc + mod_b[ci];
}

// ---- demod[b,o] = rsqrt(sum_{ci,kk}(W[o,ci,kk]*s[b,ci])^2 + 1e-8) ----
__global__ __launch_bounds__(256)
void sg2_demod_kernel(const float* __restrict__ weight,
                      const float* __restrict__ s_mod,
                      float* __restrict__ demod, int cin, int cout) {
    const int o = blockIdx.x, b = blockIdx.y;
    const float* w = weight + (size_t)o * cin * 9;
    const float* s = s_mod + b * cin;
    float acc = 0.f;
    for (int i = threadIdx.x; i < cin * 9; i += 256) {
        float ws = w[i] * s[i / 9];
        acc += ws * ws;
    }
    __shared__ float red[256];
    red[threadIdx.x] = acc;
    __syncthreads();
    for (int st = 128; st > 0; st >>= 1) {
        if (threadIdx.x < st) red[threadIdx.x] += red[threadIdx.x + st];
        __syncthreads();
    }
    if (threadIdx.x == 0) demod[b * cout + o] = rsqrtf(red[0] + 1e-8f);
}

// ---- prep: modulate (+ optional bilinear 2x upsample) + zero-pad(1) + cvt f16 ----
// out layout: [B][cin][H+2][W+2]
__global__ __launch_bounds__(256)
void sg2_prep_kernel(const float* __restrict__ in, _Float16* __restrict__ outp,
                     const float* __restrict__ s_mod,
                     int cin, int H, int W, int inH, int inW,
                     int upsample, long long in_bstride) {
    const int b = blockIdx.z;
    const int PH = H + 2, PW = W + 2;
    const int plane_sz = PH * PW;
    int idx = blockIdx.x * blockDim.x + threadIdx.x;
    if (idx >= cin * plane_sz) return;
    int ci = idx / plane_sz;
    int r  = idx - ci * plane_sz;
    int yy = r / PW, xx = r - yy * PW;
    int y = yy - 1, x = xx - 1;
    float v = 0.f;
    if ((unsigned)y < (unsigned)H && (unsigned)x < (unsigned)W) {
        const float* plane = in + (size_t)b * in_bstride + (size_t)ci * inH * inW;
        v = upsample ? bilin2x(plane, inH, inW, y, x)
                     : plane[(size_t)y * inW + x];
        v *= s_mod[b * cin + ci];
    }
    outp[(size_t)(b * cin + ci) * plane_sz + r] = (_Float16)v;
}

// ---- weight repack: [cout][cin][3][3] f32 -> [cout][9][cin] f16 (kk-major K) ----
__global__ __launch_bounds__(256)
void sg2_wpack_kernel(const float* __restrict__ w, _Float16* __restrict__ w16,
                      int cin, int cout) {
    int i = blockIdx.x * blockDim.x + threadIdx.x;
    int tot = cout * cin * 9;
    if (i >= tot) return;
    int o  = i / (cin * 9);
    int r  = i - o * cin * 9;
    int kk = r / cin;
    int ci = r - kk * cin;
    w16[i] = (_Float16)w[(size_t)o * cin * 9 + (size_t)ci * 9 + kk];
}

// ---- modulated 3x3 conv: block-cooperative implicit GEMM.
//      8 waves share one 16-pixel strip (B operand staged once in LDS via the
//      gfx1250 async global->LDS path) and cover 128 output channels.
//      M=cout, N=H*W, K=9*cin (kk-major). demod/noise/bias/leaky fused. ----
__global__ __launch_bounds__(CONV_WAVES * 32)
void sg2_modconv_kernel(const _Float16* __restrict__ xmod,   // [B][cin][H+2][W+2]
                        float* __restrict__ out,
                        const _Float16* __restrict__ w16,    // [cout][9][cin]
                        const float* __restrict__ demod,
                        const float* __restrict__ noise,
                        const float* __restrict__ noise_w,
                        const float* __restrict__ bias,
                        int cin, int cout, int H, int W) {
    // halo window of the strip, all channels: max 54 * 512 halves = 55296 B
    __shared__ _Float16 patch[27648];
    const int tid  = threadIdx.x;
    const int lane = tid & 31;
    const int wave = tid >> 5;
    const int nm = lane & 15, hi = lane >> 4;
    const int ntile = blockIdx.x, b = blockIdx.z;
    const int p0 = ntile << 4;
    const int y0 = p0 / W;                 // first row covered by this strip
    const int x0 = p0 - y0 * W;            // 0 when W<16, else multiple of 16
    const int p = p0 + nm;
    const int y = p / W, x = p - y * W;
    const int PW = W + 2;
    const size_t psz = (size_t)(H + 2) * PW;
    // window dims: rows y0-1..y1+1, cols x0-1..x0+min(W,16) (padded coords y0.., x0..)
    const int Cc = (W < 16 ? W : 16) + 2;          // 6 / 10 / 18 (even)
    const int Rr = (W < 16 ? (16 / W) : 1) + 2;    // 6 / 4  / 3
    const int wsz = Rr * Cc;                       // halves per channel (even)

    // ---- cooperative stage of the B window into LDS ----
    const _Float16* src = xmod + (size_t)b * cin * psz + (size_t)y0 * PW + x0;
    const int tot2 = (cin * wsz) >> 1;             // move 2 halves (b32) at a time
    for (int i = tid; i < tot2; i += CONV_WAVES * 32) {
        int el = i << 1;
        int ci = el / wsz;
        int r  = el - ci * wsz;
        int rr = r / Cc, cc = r - rr * Cc;         // cc even -> pair stays in row
        const _Float16* g = src + (size_t)ci * psz + (size_t)rr * PW + cc;
#if SG2_ASYNC_LDS
        __builtin_amdgcn_global_load_async_to_lds_b32(
            (as1_int*)(unsigned long long)(size_t)g,
            (as3_int*)(unsigned int)(size_t)(&patch[el]), 0, 0);
#else
        patch[el]     = g[0];
        patch[el + 1] = g[1];
#endif
    }
#if SG2_ASYNC_LDS
#  if defined(__has_builtin) && __has_builtin(__builtin_amdgcn_s_wait_asynccnt)
    __builtin_amdgcn_s_wait_asynccnt(0);
#  else
    asm volatile("s_wait_asynccnt 0x0" ::: "memory");
#  endif
#endif
    __syncthreads();

    // ---- WMMA main loop: A from global (L2-resident weights), B from LDS ----
    const int oblk = ((blockIdx.y * CONV_WAVES + wave) << 4);
    const _Float16* wrow = w16 + (size_t)(oblk + nm) * cin * 9;
    v8f c = {};
    for (int kk = 0; kk < 9; ++kk) {
        const int dy = kk / 3;
        const int dx = kk - dy * 3;
        const _Float16* abase = wrow + kk * cin;
        const _Float16* lb = patch + (size_t)(hi << 4) * wsz
                           + (size_t)(y - y0 + dy) * Cc + (x - x0 + dx);
        for (int cb = 0; cb < cin; cb += 32) {
            __builtin_prefetch(abase + cb + 32, 0, 0);   // global_prefetch_b8
            v16h a, bf;
#pragma unroll
            for (int e = 0; e < 16; ++e) {
                a[e]  = abase[cb + e + (hi << 4)];       // 16 contiguous halves
                bf[e] = lb[(size_t)(cb + e) * wsz];      // ds_load, imm offsets
            }
            c = __builtin_amdgcn_wmma_f32_16x16x32_f16(false, a, false, bf,
                                                       (short)0, c, false, false);
        }
    }
    const float nz = noise[(size_t)y * W + x] * noise_w[0];
#pragma unroll
    for (int r = 0; r < 8; ++r) {
        int o = oblk + r + (hi << 3);
        float v = c[r] * demod[b * cout + o] * 1.41421356237f + nz + bias[o];
        v = (v > 0.f) ? v : 0.2f * v;
        out[(((size_t)b * cout + o) * H + y) * W + x] = v;
    }
}

// ---- 1x1 toRGB (no demod) on the prepped modulated f16 input; fused skip add ----
__global__ __launch_bounds__(32)
void sg2_torgb_kernel(const _Float16* __restrict__ xmod,  // [B][cin][H+2][W+2]
                      float* __restrict__ out,
                      const float* __restrict__ weight,   // [3][cin] f32
                      const float* __restrict__ bias,
                      const float* __restrict__ skip,
                      int cin, int H, int W, int has_skip) {
    const int lane = threadIdx.x;
    const int nm = lane & 15, hi = lane >> 4;
    const int ntile = blockIdx.x, b = blockIdx.z;
    const int p = (ntile << 4) + nm;
    const int y = p / W, x = p - y * W;
    const int PW = W + 2;
    const size_t psz = (size_t)(H + 2) * PW;
    const _Float16* xb = xmod + ((size_t)b * cin + (size_t)(hi << 4)) * psz
                       + (size_t)(y + 1) * PW + (x + 1);
    const float wsc = (nm < 3) ? 1.f : 0.f;
    const float* wr = weight + (size_t)(nm < 3 ? nm : 0) * cin;
    v8f c = {};
    for (int cb = 0; cb < cin; cb += 32) {
        v16h a, bf;
#pragma unroll
        for (int e = 0; e < 16; ++e) {
            a[e]  = (_Float16)(wr[cb + e + (hi << 4)] * wsc);
            bf[e] = xb[(size_t)(cb + e) * psz];
        }
        c = __builtin_amdgcn_wmma_f32_16x16x32_f16(false, a, false, bf,
                                                   (short)0, c, false, false);
    }
#pragma unroll
    for (int r = 0; r < 8; ++r) {
        int o = r + (hi << 3);
        if (o < 3) {
            float v = c[r] + bias[o];
            if (has_skip) {
                const float* sp = skip + (size_t)(b * 3 + o) * (H >> 1) * (W >> 1);
                v += bilin2x(sp, H >> 1, W >> 1, y, x);
            }
            out[(size_t)(b * 3 + o) * H * W + p] = v;
        }
    }
}

extern "C" void kernel_launch(void* const* d_in, const int* in_sizes, int n_in,
                              void* d_out, int out_size, void* d_ws, size_t ws_size,
                              hipStream_t stream) {
    (void)in_sizes; (void)n_in; (void)out_size; (void)ws_size;
    int idx = 0;
    const float* styles = (const float*)d_in[idx++];
    const float* noises[13];
    for (int i = 0; i < 13; ++i) noises[i] = (const float*)d_in[idx++];
    const float* mlp_w  = (const float*)d_in[idx++];   // [8,512,512]
    const float* mlp_b  = (const float*)d_in[idx++];   // [8,512]
    const float* constp = (const float*)d_in[idx++];   // [1,512,4,4]
    struct ConvP { const float *mod_w, *mod_b, *weight, *noise_w, *bias; } cp[13];
    for (int i = 0; i < 13; ++i) {
        cp[i].mod_w   = (const float*)d_in[idx++];
        cp[i].mod_b   = (const float*)d_in[idx++];
        cp[i].weight  = (const float*)d_in[idx++];
        cp[i].noise_w = (const float*)d_in[idx++];
        cp[i].bias    = (const float*)d_in[idx++];
    }
    struct RgbP { const float *mod_w, *mod_b, *weight, *bias; } rp[7];
    for (int i = 0; i < 7; ++i) {
        rp[i].mod_w  = (const float*)d_in[idx++];
        rp[i].mod_b  = (const float*)d_in[idx++];
        rp[i].weight = (const float*)d_in[idx++];
        rp[i].bias   = (const float*)d_in[idx++];
    }

    char* wsb = (char*)d_ws;
    size_t off = 0;
    auto take = [&](size_t bytes) -> void* {
        void* pp = (void*)(wsb + off);
        off = (off + bytes + 255) & ~(size_t)255;
        return pp;
    };
    float* latent = (float*)take((size_t)BATCH * NSF * 4);
    float* smod   = (float*)take((size_t)BATCH * NSF * 4);
    float* dem    = (float*)take((size_t)BATCH * NSF * 4);
    float* skipA  = (float*)take((size_t)BATCH * 3 * 128 * 128 * 4);
    float* skipB  = (float*)take((size_t)BATCH * 3 * 128 * 128 * 4);
    _Float16* w16 = (_Float16*)take((size_t)512 * 512 * 9 * 2);            // 4.7 MB
    _Float16* xm16 = (_Float16*)take((size_t)BATCH * 512 * 258 * 258 * 2); // padded f16 input
    const size_t actBytes = (size_t)BATCH * 128 * 256 * 256 * 4;           // 256 MB max act
    float* actA = (float*)take(actBytes);
    float* actB = (float*)take(actBytes);

    sg2_style_mlp_kernel<<<1, 1024, 0, stream>>>(styles, mlp_w, mlp_b, latent);

    auto run_conv = [&](int li, const float* in, long long bstride, float* outp,
                        int cin, int cout, int res, int inres, int up) {
        int tot = BATCH * cin;
        sg2_smod_kernel<<<(tot + 255) / 256, 256, 0, stream>>>(
            latent, cp[li].mod_w, cp[li].mod_b, smod, cin);
        int pel = cin * (res + 2) * (res + 2);
        sg2_prep_kernel<<<dim3((pel + 255) / 256, 1, BATCH), 256, 0, stream>>>(
            in, xm16, smod, cin, res, res, inres, inres, up, bstride);
        int wel = cout * cin * 9;
        sg2_wpack_kernel<<<(wel + 255) / 256, 256, 0, stream>>>(
            cp[li].weight, w16, cin, cout);
        sg2_demod_kernel<<<dim3(cout, BATCH), 256, 0, stream>>>(
            cp[li].weight, smod, dem, cin, cout);
        dim3 g((res * res) / 16, cout / (16 * CONV_WAVES), BATCH);
        sg2_modconv_kernel<<<g, CONV_WAVES * 32, 0, stream>>>(
            xm16, outp, w16, dem, noises[li], cp[li].noise_w, cp[li].bias,
            cin, cout, res, res);
    };
    auto run_rgb = [&](int ti, const float* in, float* outp, const float* skip,
                       int cin, int res) {
        int tot = BATCH * cin;
        sg2_smod_kernel<<<(tot + 255) / 256, 256, 0, stream>>>(
            latent, rp[ti].mod_w, rp[ti].mod_b, smod, cin);
        int pel = cin * (res + 2) * (res + 2);
        sg2_prep_kernel<<<dim3((pel + 255) / 256, 1, BATCH), 256, 0, stream>>>(
            in, xm16, smod, cin, res, res, res, res, 0, (long long)cin * res * res);
        sg2_torgb_kernel<<<dim3((res * res) / 16, 1, BATCH), 32, 0, stream>>>(
            xm16, outp, rp[ti].weight, rp[ti].bias, skip, cin, res, res,
            skip != nullptr);
    };

    static const int CH[7] = {512, 512, 512, 512, 512, 256, 128}; // res 4..256

    run_conv(0, constp, 0, actA, 512, 512, 4, 4, 0);
    run_rgb(0, actA, skipA, nullptr, 512, 4);

    float* cur = actA;
    float* alt = actB;
    float* scur = skipA;
    float* salt = skipB;
    int cin = 512, res = 4;
    for (int j = 0; j < 6; ++j) {
        int cout = CH[j + 1];
        int nres = res * 2;
        run_conv(2 * j + 1, cur, (long long)cin * res * res, alt, cin, cout, nres, res, 1);
        { float* t = cur; cur = alt; alt = t; }
        run_conv(2 * j + 2, cur, (long long)cout * nres * nres, alt, cout, cout, nres, nres, 0);
        { float* t = cur; cur = alt; alt = t; }
        float* sdst = (j == 5) ? (float*)d_out : salt;
        run_rgb(j + 1, cur, sdst, scur, cout, nres);
        { float* t = scur; scur = salt; salt = t; }
        cin = cout; res = nres;
    }
}
